// PartDeformDecoder2_25555055411688
// MI455X (gfx1250) — matrix operand order repaired and checked by reading, not compiled, not run on hardware
//
#include <hip/hip_runtime.h>

// CDNA5 / gfx1250: wave32, WMMA f32_16x16x32_f16 path.
// Pipeline:
//   f32->f16 convert of net  -> WMMA mlp2 (K=640) with tanh, f16 activations
//   -> WMMA GEMMs (K=512, N=30000/60000) with tanh epilogue storing
//      node-major [n][b][c]
//   -> 3 GCN layers (init kernel = bias + self-loop; edge kernel = fused
//      CxC transform + norm scale + global_atomic_add_f32 scatter, with
//      wave-uniform edge metadata scalarized via readfirstlane)
//   -> final [B,N,9] assembly with *4 / *50 scales.

typedef __attribute__((ext_vector_type(16))) _Float16 v16h;
typedef __attribute__((ext_vector_type(8)))  _Float16 v8h;
typedef __attribute__((ext_vector_type(8)))  float    v8f;

static constexpr int Bn    = 64;
static constexpr int FEAT  = 512;
static constexpr int NIN   = 640;    // FEAT + SFEAT
static constexpr int HID   = 1024;   // 2*FEAT
static constexpr int Nn    = 10000;
static constexpr int En    = 160000;
static constexpr int NCL   = Nn * 3;   // 30000 cols (logr GEMM)
static constexpr int NCS   = Nn * 6;   // 60000 cols (s GEMM)
static constexpr int STRIPS_L = NCL / 16;            // 1875
static constexpr int STRIPS_S = NCS / 16;            // 3750
static constexpr int STRIPS   = STRIPS_L + STRIPS_S; // 5625

// ---------------- stage 0: convert net (f32) to f16 -------------------------
__global__ void k_cvt_net(const float* __restrict__ net, _Float16* __restrict__ neth) {
  int i = blockIdx.x * blockDim.x + threadIdx.x;
  if (i < Bn * NIN) neth[i] = (_Float16)net[i];
}

// ---------------- stage 1: net2h = f16(tanh(net @ W_mlp2)), via WMMA --------
// One wave per 16-column strip of the 1024 outputs; 4 M-tiles cover B=64.
__global__ void __launch_bounds__(128) k_mlp2_wmma(
    const _Float16* __restrict__ neth, const float* __restrict__ W,
    _Float16* __restrict__ net2h) {
  const int strip = blockIdx.x * 4 + (threadIdx.x >> 5);
  if (strip >= HID / 16) return;
  const int lane = threadIdx.x & 31;
  const int lm = lane & 15;
  const int hi = lane >> 4;
  const int ncol = strip * 16 + lm;

  v8f z = {0.f, 0.f, 0.f, 0.f, 0.f, 0.f, 0.f, 0.f};
  v8f acc[4];
#pragma unroll
  for (int t = 0; t < 4; ++t) acc[t] = z;

#pragma unroll 1
  for (int kc = 0; kc < NIN / 32; ++kc) {  // 20 chunks
    const int krow0 = kc * 32 + hi * 16;
    const float* wp = W + (size_t)krow0 * HID + ncol;
    v16h bf;
#pragma unroll
    for (int r = 0; r < 16; ++r) bf[r] = (_Float16)wp[(size_t)r * HID];
    const int k0 = kc * 32 + hi * 16;
#pragma unroll
    for (int t = 0; t < 4; ++t) {
      const int m = t * 16 + lm;
      const v8h* ap = (const v8h*)(neth + (size_t)m * NIN + k0);
      v8h lo = ap[0];
      v8h hh = ap[1];  // +8 halves
      v16h af;
#pragma unroll
      for (int i = 0; i < 8; ++i) { af[i] = lo[i]; af[8 + i] = hh[i]; }
      acc[t] = __builtin_amdgcn_wmma_f32_16x16x32_f16(
          false, af, false, bf, (short)0, acc[t], false, false);
    }
  }

  const int mb = hi * 8;
#pragma unroll
  for (int t = 0; t < 4; ++t)
#pragma unroll
    for (int r = 0; r < 8; ++r) {
      const int m = t * 16 + mb + r;
      net2h[(size_t)m * HID + ncol] = (_Float16)tanhf(acc[t][r]);
    }
}

// ---------------- degree / normalization ------------------------------------
__global__ void k_fill1(float* __restrict__ p, int n) {
  int i = blockIdx.x * blockDim.x + threadIdx.x;
  if (i < n) p[i] = 1.0f;  // self-loop contributes 1 to every degree
}
__global__ void k_degacc(const int* __restrict__ dst, float* __restrict__ deg) {
  int e = blockIdx.x * blockDim.x + threadIdx.x;
  if (e < En) atomicAdd(&deg[dst[e]], 1.0f);
}
__global__ void k_dinv(const float* __restrict__ deg, float* __restrict__ dinv) {
  int i = blockIdx.x * blockDim.x + threadIdx.x;
  if (i < Nn) dinv[i] = rsqrtf(deg[i]);
}
__global__ void k_norm(const int* __restrict__ src, const int* __restrict__ dst,
                       const float* __restrict__ dinv, float* __restrict__ norm) {
  int e = blockIdx.x * blockDim.x + threadIdx.x;
  if (e < En) norm[e] = dinv[src[e]] * dinv[dst[e]];
}

// ---------------- stage 2: big WMMA GEMMs + tanh, node-major store ----------
// One wave per 16-column strip; 4 M-tiles (64 batch rows) share each B frag,
// so weight traffic is exactly-once.
__global__ void __launch_bounds__(128) k_gemm(
    const _Float16* __restrict__ net2h,
    const float* __restrict__ Wl, const float* __restrict__ Ws,
    float* __restrict__ Xl, float* __restrict__ Xs) {
  const int strip = blockIdx.x * 4 + (threadIdx.x >> 5);
  if (strip >= STRIPS) return;
  const int lane = threadIdx.x & 31;
  const bool isL = strip < STRIPS_L;
  const int s2   = isL ? strip : strip - STRIPS_L;
  const int cb   = s2 * 16;
  const int Ncols = isL ? NCL : NCS;
  const int Koff  = isL ? 0 : FEAT;
  const float* W  = isL ? Wl : Ws;

  const int lm = lane & 15;
  const int hi = lane >> 4;
  const int ncol = cb + lm;

  v8f z = {0.f, 0.f, 0.f, 0.f, 0.f, 0.f, 0.f, 0.f};
  v8f acc[4];
#pragma unroll
  for (int t = 0; t < 4; ++t) acc[t] = z;

#pragma unroll 1
  for (int kc = 0; kc < FEAT / 32; ++kc) {
    // B fragment: 32x16; lanes 0-15 hold K rows +0..15, lanes 16-31 +16..31
    const int krow0 = kc * 32 + hi * 16;
    const float* wp = W + (size_t)krow0 * Ncols + ncol;
    v16h bf;
#pragma unroll
    for (int r = 0; r < 16; ++r) bf[r] = (_Float16)wp[(size_t)r * Ncols];

    const int k0 = Koff + kc * 32 + hi * 16;
#pragma unroll
    for (int t = 0; t < 4; ++t) {
      const int m = t * 16 + lm;
      const v8h* ap = (const v8h*)(net2h + (size_t)m * HID + k0);
      v8h lo = ap[0];
      v8h hh = ap[1];  // +8 halves
      v16h af;
#pragma unroll
      for (int i = 0; i < 8; ++i) { af[i] = lo[i]; af[8 + i] = hh[i]; }
      acc[t] = __builtin_amdgcn_wmma_f32_16x16x32_f16(
          false, af, false, bf, (short)0, acc[t], false, false);
    }
  }

  // epilogue: tanh, store node-major [n][b][c]
  const int mb = hi * 8;
#pragma unroll
  for (int t = 0; t < 4; ++t) {
#pragma unroll
    for (int r = 0; r < 8; ++r) {
      const float v = tanhf(acc[t][r]);
      const int m = t * 16 + mb + r;
      if (isL) {
        const int n = ncol / 3, c = ncol - n * 3;
        Xl[((size_t)n * Bn + m) * 3 + c] = v;
      } else {
        const int n = ncol / 6, c = ncol - n * 6;
        Xs[((size_t)n * Bn + m) * 6 + c] = v;
      }
    }
  }
}

// ---------------- GCN layers -------------------------------------------------
// out[n][b][:] = bias + (x[n][b]@W) * dinv[n]^2   (self-loop; also initializes)
template <int C>
__global__ void k_gcn_init(const float* __restrict__ x,
                           const float* __restrict__ dinv,
                           const float* __restrict__ W,
                           const float* __restrict__ bias,
                           float* __restrict__ out) {
  const int t = blockIdx.x * blockDim.x + threadIdx.x;  // t = n*64 + b
  if (t >= Nn * Bn) return;
  // n is identical across each 32-lane wave (grid exact, 64 threads/node)
  const int n = __builtin_amdgcn_readfirstlane(t >> 6);
  const float di = dinv[n];
  const float sn = di * di;
  float Wr[C * C];
#pragma unroll
  for (int i = 0; i < C * C; ++i) Wr[i] = W[i];
  float xi[C];
  const float* xp = x + (size_t)t * C;
#pragma unroll
  for (int c = 0; c < C; ++c) xi[c] = xp[c];
  float* op = out + (size_t)t * C;
#pragma unroll
  for (int d = 0; d < C; ++d) {
    float h = 0.f;
#pragma unroll
    for (int c = 0; c < C; ++c) h += xi[c] * Wr[c * C + d];
    op[d] = bias[d] + h * sn;
  }
}

// out[dst][b][:] += (x[src][b]@W) * norm[e]; 64 batch lanes per edge so the
// gather and the atomics are contiguous; e/src/dst/norm are wave-uniform.
template <int C>
__global__ void k_gcn_edge(const float* __restrict__ x,
                           const int* __restrict__ src,
                           const int* __restrict__ dst,
                           const float* __restrict__ norm,
                           const float* __restrict__ W,
                           float* __restrict__ out) {
  const long long t = (long long)blockIdx.x * blockDim.x + threadIdx.x;
  if (t >= (long long)En * Bn) return;
  const int b = (int)(t & 63);
  // e is identical across each 32-lane wave -> force scalar loads
  const int e = __builtin_amdgcn_readfirstlane((int)(t >> 6));
  const int s = src[e], d = dst[e];
  const float nr = norm[e];
  float Wr[C * C];
#pragma unroll
  for (int i = 0; i < C * C; ++i) Wr[i] = W[i];
  float xi[C];
  const float* xp = x + ((size_t)s * Bn + b) * C;
#pragma unroll
  for (int c = 0; c < C; ++c) xi[c] = xp[c];
  float* op = out + ((size_t)d * Bn + b) * C;
#pragma unroll
  for (int dd = 0; dd < C; ++dd) {
    float h = 0.f;
#pragma unroll
    for (int c = 0; c < C; ++c) h += xi[c] * Wr[c * C + dd];
    atomicAdd(&op[dd], h * nr);
  }
}

__global__ void k_tanh(float* __restrict__ p, long long n) {
  long long i = (long long)blockIdx.x * blockDim.x + threadIdx.x;
  if (i < n) p[i] = tanhf(p[i]);
}

// ---------------- final: [B,N,9] = concat(Ylogr*4, Ys*50) -------------------
__global__ void k_final(const float* __restrict__ Yl, const float* __restrict__ Ys,
                        float* __restrict__ out) {
  const int t = blockIdx.x * blockDim.x + threadIdx.x;
  if (t >= Bn * Nn) return;
  const int b = t / Nn, n = t - b * Nn;
  float* op = out + ((size_t)b * Nn + n) * 9;
  const float* lp = Yl + ((size_t)n * Bn + b) * 3;
  const float* sp = Ys + ((size_t)n * Bn + b) * 6;
  op[0] = 4.f * lp[0];
  op[1] = 4.f * lp[1];
  op[2] = 4.f * lp[2];
#pragma unroll
  for (int c = 0; c < 6; ++c) op[3 + c] = 50.f * sp[c];
}

extern "C" void kernel_launch(void* const* d_in, const int* in_sizes, int n_in,
                              void* d_out, int out_size, void* d_ws, size_t ws_size,
                              hipStream_t stream) {
  (void)in_sizes; (void)n_in; (void)out_size; (void)ws_size;
  const float* net   = (const float*)d_in[0];
  const int*   ei    = (const int*)d_in[1];   // [2,E] int32 (harness int convention)
  const float* Wmlp2 = (const float*)d_in[2];
  const float* Wlogr = (const float*)d_in[3];
  const float* Wsbig = (const float*)d_in[4];
  const float* W1l   = (const float*)d_in[5];
  const float* b1l   = (const float*)d_in[6];
  const float* W1s   = (const float*)d_in[7];
  const float* b1s   = (const float*)d_in[8];
  const float* W4s   = (const float*)d_in[9];
  const float* b4s   = (const float*)d_in[10];
  float* out = (float*)d_out;
  float* ws  = (float*)d_ws;

  const int* src = ei;
  const int* dst = ei + En;

  // workspace layout (float offsets, all 16B aligned); total ~47.0 MB
  _Float16* net2h = (_Float16*)ws;            // 65536 halves = 32768 floats
  _Float16* neth  = (_Float16*)(ws + 32768);  // 40960 halves = 20480 floats
  float* deg  = ws + 53248;                   // 10000
  float* dinv = ws + 63248;                   // 10000
  float* norm = ws + 73248;                   // 160000
  float* Xl   = ws + 233248;                  // 10000*64*3 = 1,920,000
  float* Xs   = ws + 2153248;                 // 10000*64*6 = 3,840,000
  float* Yl   = ws + 5993248;                 // 1,920,000
  float* T    = ws + 7913248;                 // 3,840,000 (end 11,753,248 floats)
  float* Ys   = Xs;                           // Xs dead after tanh(T); reuse

  const int edgeBlocks = (int)(((long long)En * Bn + 255) / 256);  // 40000

  k_cvt_net<<<(Bn * NIN + 255) / 256, 256, 0, stream>>>(net, neth);
  k_mlp2_wmma<<<(HID / 16 + 3) / 4, 128, 0, stream>>>(neth, Wmlp2, net2h);

  k_fill1<<<(Nn + 255) / 256, 256, 0, stream>>>(deg, Nn);
  k_degacc<<<(En + 255) / 256, 256, 0, stream>>>(dst, deg);
  k_dinv<<<(Nn + 255) / 256, 256, 0, stream>>>(deg, dinv);
  k_norm<<<(En + 255) / 256, 256, 0, stream>>>(src, dst, dinv, norm);

  k_gemm<<<(STRIPS + 3) / 4, 128, 0, stream>>>(net2h, Wlogr, Wsbig, Xl, Xs);

  // layer 1: Yl = gcn(Xl, W1_logr, b1_logr)
  k_gcn_init<3><<<(Nn * Bn + 255) / 256, 256, 0, stream>>>(Xl, dinv, W1l, b1l, Yl);
  k_gcn_edge<3><<<edgeBlocks, 256, 0, stream>>>(Xl, src, dst, norm, W1l, Yl);
  // layer 2: T = tanh(gcn(Xs, W1_s, b1_s))
  k_gcn_init<6><<<(Nn * Bn + 255) / 256, 256, 0, stream>>>(Xs, dinv, W1s, b1s, T);
  k_gcn_edge<6><<<edgeBlocks, 256, 0, stream>>>(Xs, src, dst, norm, W1s, T);
  k_tanh<<<(int)(((long long)Nn * Bn * 6 + 255) / 256), 256, 0, stream>>>(T, (long long)Nn * Bn * 6);
  // layer 3: Ys = gcn(T, W4_s, b4_s)
  k_gcn_init<6><<<(Nn * Bn + 255) / 256, 256, 0, stream>>>(T, dinv, W4s, b4s, Ys);
  k_gcn_edge<6><<<edgeBlocks, 256, 0, stream>>>(T, src, dst, norm, W4s, Ys);

  k_final<<<(Bn * Nn + 255) / 256, 256, 0, stream>>>(Yl, Ys, out);
}
